// EncoderLayer_1967095021846
// MI455X (gfx1250) — matrix operand order repaired and checked by reading, not compiled, not run on hardware
//
#include <hip/hip_runtime.h>
#include <hip/hip_bf16.h>

typedef __attribute__((ext_vector_type(16))) __bf16 v16bf;
typedef __attribute__((ext_vector_type(8)))  __bf16 v8bf;
typedef __attribute__((ext_vector_type(8)))  float  v8f;
typedef __attribute__((ext_vector_type(4)))  float  v4f;

#define BDIM 4
#define TDIM 1024
#define DDIM 512
#define HDIM 8
#define FDIM 2048
#define DKV  64
#define KCONV 31

// Build a v16bf fragment from two contiguous 16-byte LDS chunks.
__device__ __forceinline__ v16bf frag16(const __bf16* p0, const __bf16* p1)
{
    v8bf lo = *(const v8bf*)p0;
    v8bf hi = *(const v8bf*)p1;
    return __builtin_shufflevector(lo, hi, 0, 1, 2, 3, 4, 5, 6, 7,
                                           8, 9, 10, 11, 12, 13, 14, 15);
}

// ---------------------------------------------------------------------------
// LayerNorm: one 256-thread block per row of 512
// ---------------------------------------------------------------------------
__global__ __launch_bounds__(256)
void ln_kernel(const float* __restrict__ x, const float* __restrict__ g,
               const float* __restrict__ bb, float* __restrict__ out)
{
    int row = blockIdx.x;
    const float* xr = x + (long long)row * DDIM;
    float* orow = out + (long long)row * DDIM;
    int tid = threadIdx.x;
    float v0 = xr[tid], v1 = xr[tid + 256];
    __shared__ float red[256];
    red[tid] = v0 + v1;
    __syncthreads();
    for (int o = 128; o > 0; o >>= 1) {
        if (tid < o) red[tid] += red[tid + o];
        __syncthreads();
    }
    float mean = red[0] * (1.0f / DDIM);
    __syncthreads();
    float d0 = v0 - mean, d1 = v1 - mean;
    red[tid] = d0 * d0 + d1 * d1;
    __syncthreads();
    for (int o = 128; o > 0; o >>= 1) {
        if (tid < o) red[tid] += red[tid + o];
        __syncthreads();
    }
    float inv = rsqrtf(red[0] * (1.0f / DDIM) + 1e-5f);
    orow[tid]       = d0 * inv * g[tid]       + bb[tid];
    orow[tid + 256] = d1 * inv * g[tid + 256] + bb[tid + 256];
}

// ---------------------------------------------------------------------------
// Generic batched GEMM via v_wmma_f32_16x16x32_bf16.
// Block tile 128x64, 128 threads = 4 waves; each wave owns 32 rows x 64 cols
// (8 accumulators). Double-buffered LDS tiles; staging goes through registers
// so all global loads of a K-slab issue back-to-back and are consumed after
// the WMMA block (load latency hidden behind compute, 1 barrier / K-step).
//   As[m][k]  row-major  (A frag = As[m][half*8], As[m][16+half*8])
//   Bst[n][k] N-major    (B frag = Bst[n][half*8], Bst[n][16+half*8])
// Batch offset = i1*s?1 + i2*s?2 with blockIdx.z = i1*nB2 + i2.
// Out = resScale*Res + cScale*act(A@B + bias)
// ---------------------------------------------------------------------------
__global__ __launch_bounds__(128)
void gemm_bf16_wmma(const float* __restrict__ A, long long sA1, long long sA2, int lda,
                    const float* __restrict__ Bm, long long sB1, long long sB2, int ldb, int transB,
                    const float* __restrict__ bias,
                    const float* __restrict__ Res, long long sR1, long long sR2, int ldr,
                    float* __restrict__ Out, long long sC1, long long sC2, int ldc,
                    int M, int N, int K, int nB2,
                    float cScale, float resScale, int relu)
{
    int bz = blockIdx.z;
    long long i1 = bz / nB2, i2 = bz % nB2;
    A  += i1 * sA1 + i2 * sA2;
    Bm += i1 * sB1 + i2 * sB2;
    Out += i1 * sC1 + i2 * sC2;
    if (Res) Res += i1 * sR1 + i2 * sR2;

    int m0 = blockIdx.y * 128;
    int n0 = blockIdx.x * 64;

    __shared__ __bf16 As[2][128][40];   // 2 x 128x32, padded rows (80B, 16B-aligned)
    __shared__ __bf16 Bst[2][64][40];   // 2 x (32x64 stored N-major)

    int tid  = threadIdx.x;
    int lane = tid & 31;
    int wave = tid >> 5;            // 0..3 -> 32-row slice
    int half = lane >> 4;
    int lm   = lane & 15;

    // Per-thread staging coordinates (fixed across K-steps)
    int arow = tid >> 3;            // 0..15 base row (stepped by 16 over 8 chunks)
    int acol = (tid & 7) * 4;       // 0..28
    int bcT_c  = tid >> 3;          // transB: n (stepped by 16)
    int bcT_r4 = (tid & 7) * 4;     // transB: k
    int bcN_r  = tid >> 4;          // !transB: k (stepped by 8)
    int bcN_c4 = (tid & 15) * 4;    // !transB: n

    v4f ra[8], rb[4];

    auto loadA = [&](int k0) {
#pragma unroll
        for (int u = 0; u < 8; ++u)
            ra[u] = *(const v4f*)&A[(long long)(m0 + arow + u * 16) * lda + k0 + acol];
    };
    auto loadB = [&](int k0) {
        if (transB) {
#pragma unroll
            for (int u = 0; u < 4; ++u)
                rb[u] = *(const v4f*)&Bm[(long long)(n0 + bcT_c + u * 16) * ldb + k0 + bcT_r4];
        } else {
#pragma unroll
            for (int u = 0; u < 4; ++u)
                rb[u] = *(const v4f*)&Bm[(long long)(k0 + bcN_r + u * 8) * ldb + n0 + bcN_c4];
        }
    };
    auto storeTiles = [&](int buf) {
#pragma unroll
        for (int u = 0; u < 8; ++u) {
            As[buf][arow + u * 16][acol + 0] = (__bf16)ra[u].x;
            As[buf][arow + u * 16][acol + 1] = (__bf16)ra[u].y;
            As[buf][arow + u * 16][acol + 2] = (__bf16)ra[u].z;
            As[buf][arow + u * 16][acol + 3] = (__bf16)ra[u].w;
        }
        if (transB) {
#pragma unroll
            for (int u = 0; u < 4; ++u) {
                Bst[buf][bcT_c + u * 16][bcT_r4 + 0] = (__bf16)rb[u].x;
                Bst[buf][bcT_c + u * 16][bcT_r4 + 1] = (__bf16)rb[u].y;
                Bst[buf][bcT_c + u * 16][bcT_r4 + 2] = (__bf16)rb[u].z;
                Bst[buf][bcT_c + u * 16][bcT_r4 + 3] = (__bf16)rb[u].w;
            }
        } else {
#pragma unroll
            for (int u = 0; u < 4; ++u) {
                Bst[buf][bcN_c4 + 0][bcN_r + u * 8] = (__bf16)rb[u].x;
                Bst[buf][bcN_c4 + 1][bcN_r + u * 8] = (__bf16)rb[u].y;
                Bst[buf][bcN_c4 + 2][bcN_r + u * 8] = (__bf16)rb[u].z;
                Bst[buf][bcN_c4 + 3][bcN_r + u * 8] = (__bf16)rb[u].w;
            }
        }
    };

    v8f zero = {0, 0, 0, 0, 0, 0, 0, 0};
    v8f acc[2][4];
#pragma unroll
    for (int mi = 0; mi < 2; ++mi)
#pragma unroll
        for (int j = 0; j < 4; ++j) acc[mi][j] = zero;

    int nK = K >> 5;
    loadA(0); loadB(0);
    storeTiles(0);
    __syncthreads();

    for (int kt = 0; kt < nK; ++kt) {
        int buf = kt & 1;
        if (kt + 1 < nK) {          // issue next slab's global loads now
            loadA((kt + 1) << 5);
            loadB((kt + 1) << 5);
        }

        // ---- 8 WMMAs on current buffer ----
        v16bf af[2];
#pragma unroll
        for (int mi = 0; mi < 2; ++mi) {
            const __bf16* ar = As[buf][wave * 32 + mi * 16 + lm];
            af[mi] = frag16(&ar[half * 8], &ar[16 + half * 8]);
        }
#pragma unroll
        for (int j = 0; j < 4; ++j) {
            const __bf16* br = Bst[buf][j * 16 + lm];
            v16bf bfr = frag16(&br[half * 8], &br[16 + half * 8]);
            acc[0][j] = __builtin_amdgcn_wmma_f32_16x16x32_bf16(
                false, af[0], false, bfr, (short)0, acc[0][j], false, false);
            acc[1][j] = __builtin_amdgcn_wmma_f32_16x16x32_bf16(
                false, af[1], false, bfr, (short)0, acc[1][j], false, false);
        }

        if (kt + 1 < nK) {
            storeTiles(buf ^ 1);    // other buffer: no race with lagging readers
            __syncthreads();
        }
    }

    // ---- epilogue: C layout (ISA 7.12.2): N = lm, M = r + 8*half ----
#pragma unroll
    for (int mi = 0; mi < 2; ++mi) {
#pragma unroll
        for (int j = 0; j < 4; ++j) {
            int n = n0 + j * 16 + lm;
            if (n >= N) continue;
            float bv = bias ? bias[n] : 0.0f;
#pragma unroll
            for (int r = 0; r < 8; ++r) {
                int m = m0 + wave * 32 + mi * 16 + r + 8 * half;
                if (m >= M) continue;
                float c = acc[mi][j][r] + bv;
                if (relu) c = fmaxf(c, 0.0f);
                float o = cScale * c;
                if (Res) o += resScale * Res[(long long)m * ldr + n];
                Out[(long long)m * ldc + n] = o;
            }
        }
    }
}

// ---------------------------------------------------------------------------
// Relative-position term: for fixed t, scores[bh, t, s] += Q_t(32x64) @ pos_k[t]^T
// 256 threads = 8 waves, each owns one 16x16 tile of the 32x64 output.
// Single-shot staging (one tile), contiguous-fragment LDS layouts.
// ---------------------------------------------------------------------------
__global__ __launch_bounds__(256)
void posbias_wmma(const float* __restrict__ q, const float* __restrict__ pos_k,
                  float* __restrict__ scores)
{
    int t  = blockIdx.y;
    int s0 = blockIdx.x * 64;

    __shared__ __bf16 As[32][72];    // (b*8+h) x k, padded (144B rows)
    __shared__ __bf16 Bst[64][72];   // s-major: Bst[n][k]

    int tid = threadIdx.x;
    // register staging: issue all global loads, then convert+store
    v4f ra[2], rb[4];
#pragma unroll
    for (int u = 0; u < 2; ++u) {
        int i = tid + u * 256;
        int r = i >> 4, c4 = (i & 15) * 4;
        int b = r >> 3, h = r & 7;
        ra[u] = *(const v4f*)&q[((long long)(b * TDIM + t)) * DDIM + h * DKV + c4];
    }
#pragma unroll
    for (int u = 0; u < 4; ++u) {
        int i = tid + u * 256;
        int n = i >> 4, k4 = (i & 15) * 4;
        rb[u] = *(const v4f*)&pos_k[((long long)t * TDIM + (s0 + n)) * DKV + k4];
    }
#pragma unroll
    for (int u = 0; u < 2; ++u) {
        int i = tid + u * 256;
        int r = i >> 4, c4 = (i & 15) * 4;
        As[r][c4 + 0] = (__bf16)ra[u].x;
        As[r][c4 + 1] = (__bf16)ra[u].y;
        As[r][c4 + 2] = (__bf16)ra[u].z;
        As[r][c4 + 3] = (__bf16)ra[u].w;
    }
#pragma unroll
    for (int u = 0; u < 4; ++u) {
        int i = tid + u * 256;
        int n = i >> 4, k4 = (i & 15) * 4;
        Bst[n][k4 + 0] = (__bf16)rb[u].x;
        Bst[n][k4 + 1] = (__bf16)rb[u].y;
        Bst[n][k4 + 2] = (__bf16)rb[u].z;
        Bst[n][k4 + 3] = (__bf16)rb[u].w;
    }
    __syncthreads();

    int lane = tid & 31, wave = tid >> 5;
    int mt = wave >> 2;          // 0..1  (rows of 16)
    int nt = wave & 3;           // 0..3  (cols of 16)
    int half = lane >> 4, lm = lane & 15;

    v8f acc = {0, 0, 0, 0, 0, 0, 0, 0};
#pragma unroll
    for (int k0 = 0; k0 < 64; k0 += 32) {
        const __bf16* ar = As[mt * 16 + lm];
        const __bf16* br = Bst[nt * 16 + lm];
        v16bf af = frag16(&ar[k0 + half * 8], &ar[k0 + 16 + half * 8]);
        v16bf bfr = frag16(&br[k0 + half * 8], &br[k0 + 16 + half * 8]);
        acc = __builtin_amdgcn_wmma_f32_16x16x32_bf16(
            false, af, false, bfr, (short)0, acc, false, false);
    }

#pragma unroll
    for (int r = 0; r < 8; ++r) {
        int m = mt * 16 + r + 8 * half;          // bh index
        int n = nt * 16 + lm;
        long long idx = (long long)m * TDIM * TDIM + (long long)t * TDIM + s0 + n;
        scores[idx] += acc[r];
    }
}

// ---------------------------------------------------------------------------
// Masked softmax over s (row of 1024); applies 1/sqrt(DK), zeroes masked lanes.
// ---------------------------------------------------------------------------
__global__ __launch_bounds__(256)
void softmax_kernel(float* __restrict__ scores, const int* __restrict__ mask)
{
    int row = blockIdx.x;              // b*H*T + h*T + t
    int t  = row & (TDIM - 1);
    int bh = row >> 10;
    int b  = bh >> 3;
    float* sr = scores + (long long)row * TDIM;
    const int* mr = mask + ((long long)b * TDIM + t) * TDIM;
    int tid = threadIdx.x;
    __shared__ float red[256];

    float v[4]; int mk[4];
    float mx = -3.4e38f;
#pragma unroll
    for (int i = 0; i < 4; ++i) {
        int s = tid + i * 256;
        mk[i] = mr[s];
        v[i] = (mk[i] == 0) ? -3.4e38f : sr[s] * 0.125f;
        mx = fmaxf(mx, v[i]);
    }
    red[tid] = mx; __syncthreads();
    for (int o = 128; o > 0; o >>= 1) {
        if (tid < o) red[tid] = fmaxf(red[tid], red[tid + o]);
        __syncthreads();
    }
    mx = red[0]; __syncthreads();

    float sum = 0.0f;
#pragma unroll
    for (int i = 0; i < 4; ++i) {
        float e = (mk[i] == 0) ? 0.0f : __expf(v[i] - mx);
        v[i] = e; sum += e;
    }
    red[tid] = sum; __syncthreads();
    for (int o = 128; o > 0; o >>= 1) {
        if (tid < o) red[tid] += red[tid + o];
        __syncthreads();
    }
    float invs = (red[0] > 0.0f) ? 1.0f / red[0] : 0.0f;
#pragma unroll
    for (int i = 0; i < 4; ++i)
        sr[tid + i * 256] = v[i] * invs;
}

// ---------------------------------------------------------------------------
// GLU pointwise: y = (w0*y+b0) * sigmoid(w1*y+b1)   (in-place, elementwise)
// ---------------------------------------------------------------------------
__global__ __launch_bounds__(256)
void glu_kernel(float* __restrict__ y, const float* __restrict__ pw1_w,
                const float* __restrict__ pw1_b)
{
    long long i = (long long)blockIdx.x * 256 + threadIdx.x;
    float v = y[i];
    float a = pw1_w[0] * v + pw1_b[0];
    float g = pw1_w[1] * v + pw1_b[1];
    y[i] = a * (1.0f / (1.0f + __expf(-g)));
}

// ---------------------------------------------------------------------------
// Depthwise conv (K=31) along T + bias + ReLU + pointwise(1->1) + residual
// xb[b,t,d] += pw2_w*relu(conv(y)[b,t,d]+dw_b[d]) + pw2_b
// ---------------------------------------------------------------------------
__global__ __launch_bounds__(256)
void dwconv_kernel(const float* __restrict__ y, const float* __restrict__ dw_w,
                   const float* __restrict__ dw_b, const float* __restrict__ pw2_w,
                   const float* __restrict__ pw2_b, float* __restrict__ xb)
{
    long long i = (long long)blockIdx.x * 256 + threadIdx.x;   // over B*T*D
    int d = (int)(i & (DDIM - 1));
    long long bt = i >> 9;
    int t = (int)(bt & (TDIM - 1));
    int b = (int)(bt >> 10);
    float acc = 0.0f;
#pragma unroll
    for (int kk = 0; kk < KCONV; ++kk) {
        int tt = t + kk - (KCONV - 1) / 2;
        if (tt >= 0 && tt < TDIM)
            acc += y[((long long)(b * TDIM + tt)) * DDIM + d] * dw_w[d * KCONV + kk];
    }
    acc += dw_b[d];
    acc = fmaxf(acc, 0.0f);
    xb[i] = xb[i] + pw2_w[0] * acc + pw2_b[0];
}

// ---------------------------------------------------------------------------
// Host-side orchestration
// ---------------------------------------------------------------------------
extern "C" void kernel_launch(void* const* d_in, const int* in_sizes, int n_in,
                              void* d_out, int out_size, void* d_ws, size_t ws_size,
                              hipStream_t stream)
{
    const float* x_in    = (const float*)d_in[0];
    const float* pos_k   = (const float*)d_in[1];
    const int*   mask    = (const int*)  d_in[2];
    const float* fin_lg  = (const float*)d_in[3];
    const float* fin_lb  = (const float*)d_in[4];
    const float* fin_w1  = (const float*)d_in[5];
    const float* fin_b1  = (const float*)d_in[6];
    const float* fin_w2  = (const float*)d_in[7];
    const float* fin_b2  = (const float*)d_in[8];
    const float* att_lg  = (const float*)d_in[9];
    const float* att_lb  = (const float*)d_in[10];
    const float* wq = (const float*)d_in[11]; const float* bq = (const float*)d_in[12];
    const float* wk = (const float*)d_in[13]; const float* bk = (const float*)d_in[14];
    const float* wv = (const float*)d_in[15]; const float* bv = (const float*)d_in[16];
    const float* wo = (const float*)d_in[17]; const float* bo = (const float*)d_in[18];
    const float* cv_lg = (const float*)d_in[19];
    const float* cv_lb = (const float*)d_in[20];
    const float* pw1_w = (const float*)d_in[21];
    const float* pw1_b = (const float*)d_in[22];
    const float* dw_w  = (const float*)d_in[23];
    const float* dw_b  = (const float*)d_in[24];
    const float* pw2_w = (const float*)d_in[25];
    const float* pw2_b = (const float*)d_in[26];
    const float* fo_lg = (const float*)d_in[27];
    const float* fo_lb = (const float*)d_in[28];
    const float* fo_w1 = (const float*)d_in[29];
    const float* fo_b1 = (const float*)d_in[30];
    const float* fo_w2 = (const float*)d_in[31];
    const float* fo_b2 = (const float*)d_in[32];
    const float* fl_lg = (const float*)d_in[33];
    const float* fl_lb = (const float*)d_in[34];

    float* out = (float*)d_out;

    // workspace layout (floats)
    const long long NXD = (long long)BDIM * TDIM * DDIM;     // 2,097,152
    float* ws = (float*)d_ws;
    float* xb     = ws;                       // running x
    float* ybuf   = xb + NXD;
    float* hbuf   = ybuf + NXD;               // B*T*F
    float* qb     = hbuf + (long long)BDIM * TDIM * FDIM;
    float* kb     = qb + NXD;
    float* vb     = kb + NXD;
    float* ctx    = vb + NXD;
    float* scores = ctx + NXD;                // B*H*T*T fp32

    const int ROWS = BDIM * TDIM;             // 4096
    const long long sTD = (long long)TDIM * DDIM;       // batch b stride in (B,T,D)
    const long long sTT = (long long)TDIM * TDIM;

    // ---- macaron FFN (half residual) ----
    ln_kernel<<<ROWS, 256, 0, stream>>>(x_in, fin_lg, fin_lb, ybuf);
    gemm_bf16_wmma<<<dim3(FDIM/64, ROWS/128, 1), 128, 0, stream>>>(
        ybuf, 0, 0, DDIM,  fin_w1, 0, 0, FDIM, 0,  fin_b1,
        nullptr, 0, 0, 0,  hbuf, 0, 0, FDIM,
        ROWS, FDIM, DDIM, 1,  1.0f, 0.0f, 1);
    gemm_bf16_wmma<<<dim3(DDIM/64, ROWS/128, 1), 128, 0, stream>>>(
        hbuf, 0, 0, FDIM,  fin_w2, 0, 0, DDIM, 0,  fin_b2,
        x_in, 0, 0, DDIM,  xb, 0, 0, DDIM,
        ROWS, DDIM, FDIM, 1,  0.5f, 1.0f, 0);

    // ---- attention ----
    ln_kernel<<<ROWS, 256, 0, stream>>>(xb, att_lg, att_lb, ybuf);
    gemm_bf16_wmma<<<dim3(DDIM/64, ROWS/128, 1), 128, 0, stream>>>(
        ybuf, 0, 0, DDIM,  wq, 0, 0, DDIM, 0,  bq,
        nullptr, 0, 0, 0,  qb, 0, 0, DDIM,  ROWS, DDIM, DDIM, 1, 1.0f, 0.0f, 0);
    gemm_bf16_wmma<<<dim3(DDIM/64, ROWS/128, 1), 128, 0, stream>>>(
        ybuf, 0, 0, DDIM,  wk, 0, 0, DDIM, 0,  bk,
        nullptr, 0, 0, 0,  kb, 0, 0, DDIM,  ROWS, DDIM, DDIM, 1, 1.0f, 0.0f, 0);
    gemm_bf16_wmma<<<dim3(DDIM/64, ROWS/128, 1), 128, 0, stream>>>(
        ybuf, 0, 0, DDIM,  wv, 0, 0, DDIM, 0,  bv,
        nullptr, 0, 0, 0,  vb, 0, 0, DDIM,  ROWS, DDIM, DDIM, 1, 1.0f, 0.0f, 0);

    // scores[bh] = q[bh] @ k[bh]^T   (batched over 32 (b,h); strided (B,T,D) views)
    gemm_bf16_wmma<<<dim3(TDIM/64, TDIM/128, BDIM*HDIM), 128, 0, stream>>>(
        qb, sTD, DKV, DDIM,   kb, sTD, DKV, DDIM, 1,  nullptr,
        nullptr, 0, 0, 0,  scores, HDIM*sTT, sTT, TDIM,
        TDIM, TDIM, DKV, HDIM,  1.0f, 0.0f, 0);

    // scores += rel-pos term (per-t GEMM: pos_k[t] read once for all 32 bh)
    posbias_wmma<<<dim3(TDIM/64, TDIM, 1), 256, 0, stream>>>(qb, pos_k, scores);

    // masked softmax (applies 1/sqrt(DK))
    softmax_kernel<<<BDIM*HDIM*TDIM, 256, 0, stream>>>(scores, mask);

    // ctx[bh] = attn[bh] @ v[bh] -> stored directly into (B,T,D) layout
    gemm_bf16_wmma<<<dim3(1, TDIM/128, BDIM*HDIM), 128, 0, stream>>>(
        scores, HDIM*sTT, sTT, TDIM,   vb, sTD, DKV, DDIM, 0,  nullptr,
        nullptr, 0, 0, 0,  ctx, sTD, DKV, DDIM,
        TDIM, DKV, TDIM, HDIM,  1.0f, 0.0f, 0);

    // x += ctx @ wo + bo
    gemm_bf16_wmma<<<dim3(DDIM/64, ROWS/128, 1), 128, 0, stream>>>(
        ctx, 0, 0, DDIM,  wo, 0, 0, DDIM, 0,  bo,
        xb, 0, 0, DDIM,  xb, 0, 0, DDIM,  ROWS, DDIM, DDIM, 1, 1.0f, 1.0f, 0);

    // ---- conv module ----
    ln_kernel<<<ROWS, 256, 0, stream>>>(xb, cv_lg, cv_lb, ybuf);
    glu_kernel<<<(int)(NXD / 256), 256, 0, stream>>>(ybuf, pw1_w, pw1_b);
    dwconv_kernel<<<(int)(NXD / 256), 256, 0, stream>>>(ybuf, dw_w, dw_b, pw2_w, pw2_b, xb);

    // ---- output FFN (half residual) ----
    ln_kernel<<<ROWS, 256, 0, stream>>>(xb, fo_lg, fo_lb, ybuf);
    gemm_bf16_wmma<<<dim3(FDIM/64, ROWS/128, 1), 128, 0, stream>>>(
        ybuf, 0, 0, DDIM,  fo_w1, 0, 0, FDIM, 0,  fo_b1,
        nullptr, 0, 0, 0,  hbuf, 0, 0, FDIM,
        ROWS, FDIM, DDIM, 1,  1.0f, 0.0f, 1);
    gemm_bf16_wmma<<<dim3(DDIM/64, ROWS/128, 1), 128, 0, stream>>>(
        hbuf, 0, 0, FDIM,  fo_w2, 0, 0, DDIM, 0,  fo_b2,
        xb, 0, 0, DDIM,  xb, 0, 0, DDIM,
        ROWS, DDIM, FDIM, 1,  0.5f, 1.0f, 0);

    // ---- final LN -> d_out[0] ----
    ln_kernel<<<ROWS, 256, 0, stream>>>(xb, fl_lg, fl_lb, out);

    // tuple outputs: echo pos_k and mask into d_out
    const long long OUT0 = NXD;                                   // 2,097,152 floats
    const long long POSN = (long long)TDIM * TDIM * DKV;          // 67,108,864 floats
    const long long MSKN = (long long)BDIM * TDIM * TDIM;         // 4,194,304 ints
    hipMemcpyAsync(out + OUT0, pos_k, POSN * sizeof(float),
                   hipMemcpyDeviceToDevice, stream);
    hipMemcpyAsync(out + OUT0 + POSN, mask, MSKN * sizeof(int),
                   hipMemcpyDeviceToDevice, stream);
    (void)in_sizes; (void)n_in; (void)out_size; (void)ws_size;
}
